// ScaleDotProduct_30829275250742
// MI455X (gfx1250) — matrix operand order repaired and checked by reading, not compiled, not run on hardware
//
#include <hip/hip_runtime.h>

// Flash-attention B=1,H=16,S=4096,D=64 (f32 in/out) on gfx1250.
// 18 v_wmma_f32_16x16x32_f16 per 64-key iteration (8 QK^T, 8 PV, 2 row-sum).
// Row sums via WMMA against an all-ones B; row max via packed-f16
// v_permlane16_b32 xor-shuffles + v_pk_max (2 rows per register).
// Key-permuted P/V layouts give b64 vector DS ops; split
// s_barrier_signal/-wait overlaps global prefetch with the barrier.

typedef __attribute__((ext_vector_type(16))) _Float16 v16h;
typedef __attribute__((ext_vector_type(8)))  _Float16 v8h;
typedef __attribute__((ext_vector_type(4)))  _Float16 v4h;
typedef __attribute__((ext_vector_type(2)))  _Float16 v2h;
typedef __attribute__((ext_vector_type(8)))  float    v8f;

#define H_  16
#define S_  4096
#define D_  64
#define KBLK 64
#define NB   (S_ / KBLK)
#define WAVES 8
#define ROWS_PER_BLOCK 128

static __device__ __forceinline__ v16h cat8(v8h a, v8h b) {
    return __builtin_shufflevector(a, b, 0,1,2,3,4,5,6,7,8,9,10,11,12,13,14,15);
}

// xor-permute within each 16-lane row via v_permlane16_b32 (static selects)
static __device__ __forceinline__ v2h plx16h2(v2h x, unsigned s0, unsigned s1) {
    unsigned u = __builtin_bit_cast(unsigned, x);
    unsigned r = __builtin_amdgcn_permlane16(u, u, s0, s1, false, false);
    return __builtin_bit_cast(v2h, r);
}

__global__ __launch_bounds__(256)
void fa_fwd_gfx1250(const float* __restrict__ q, const float* __restrict__ k,
                    const float* __restrict__ v, float* __restrict__ o)
{
    // ping-pong K tile [key][dim]; V^T tile [dim][permuted key j]; per-wave P scratch
    __shared__ __align__(16) _Float16 sK [2][KBLK][D_ + 8];     // row stride 144B
    __shared__ __align__(16) _Float16 sVt[2][D_][KBLK + 8];     // row stride 144B
    __shared__ __align__(16) _Float16 sP [WAVES][16][KBLK + 8]; // row stride 144B

    const int h    = blockIdx.y;
    const int qb   = blockIdx.x * ROWS_PER_BLOCK;
    const int tid  = threadIdx.x;
    const int wv   = tid >> 5;
    const int lane = tid & 31;
    const int half = lane >> 4;
    const int ln   = lane & 15;
    const int kg   = half * 8;

    // ---- Q tile -> two A fragments; fold 1/sqrt(D)*log2(e) into Q ----
    const float scale = 0.125f * 1.44269504088896340736f;
    const int   qrow  = qb + wv * 16 + ln;
    const float* qp   = q + ((size_t)(h * S_ + qrow)) * D_;
    v16h qa0, qa1, bones;
    #pragma unroll
    for (int j = 0; j < 8; ++j) {
        qa0[j]     = (_Float16)(qp[     kg + j] * scale);
        qa0[j + 8] = (_Float16)(qp[16 + kg + j] * scale);
        qa1[j]     = (_Float16)(qp[32 + kg + j] * scale);
        qa1[j + 8] = (_Float16)(qp[48 + kg + j] * scale);
    }
    #pragma unroll
    for (int j = 0; j < 16; ++j) bones[j] = (_Float16)1.0f;

    float mrow[8];
    v8f acc[5];                      // [0..3]: O dim groups; [4]: row sums
    #pragma unroll
    for (int i = 0; i < 8; ++i) mrow[i] = -__builtin_inff();
    #pragma unroll
    for (int g = 0; g < 5; ++g) acc[g] = {};

    // ---- staging maps ----
    const int srow = tid >> 2;            // K: key row, 16 dims
    const int scol = (tid & 3) * 16;
    const float* kbase = k + ((size_t)(h * S_ + srow)) * D_ + scol;
    const int vc = tid & 15;              // V: keys {vc,vc+16,vc+32,vc+48} x dims [vd,vd+4)
    const int vd = (tid >> 4) * 4;
    const float* vbase = v + ((size_t)(h * S_ + vc)) * D_ + vd;

    float4 krf[4], vrf[4];
    #pragma unroll
    for (int j = 0; j < 4; ++j) {
        krf[j] = *(const float4*)(kbase + j * 4);
        vrf[j] = *(const float4*)(vbase + (size_t)j * 16 * D_);
    }
    {   // stage tile 0 into buffer 0
        v8h a, b;
        const float* kr = (const float*)krf;
        #pragma unroll
        for (int j = 0; j < 8; ++j) { a[j] = (_Float16)kr[j]; b[j] = (_Float16)kr[8 + j]; }
        *(v8h*)&sK[0][srow][scol]     = a;
        *(v8h*)&sK[0][srow][scol + 8] = b;
        #pragma unroll
        for (int di = 0; di < 4; ++di) {
            v4h p;
            #pragma unroll
            for (int m = 0; m < 4; ++m) p[m] = (_Float16)((const float*)&vrf[m])[di];
            *(v4h*)&sVt[0][vd + di][4 * vc] = p;
        }
    }
    asm volatile("s_wait_dscnt 0x0\n\ts_barrier_signal -1" ::: "memory");

    for (int kb = 0; kb < NB; ++kb) {
        const int  cur  = kb & 1;
        const bool more = (kb + 1) < NB;

        // next tile's global loads issue before the barrier wait (latency overlap)
        if (more) {
            const size_t off = (size_t)(kb + 1) * KBLK * D_;
            #pragma unroll
            for (int j = 0; j < 4; ++j) {
                krf[j] = *(const float4*)(kbase + off + j * 4);
                vrf[j] = *(const float4*)(vbase + off + (size_t)j * 16 * D_);
            }
        }
        asm volatile("s_barrier_wait -1" ::: "memory");

        // ---- S = Q K^T : four 16x16 C fragments ----
        v8f sc[4];
        #pragma unroll
        for (int nb = 0; nb < 4; ++nb) {
            const int key = nb * 16 + ln;
            v16h b0 = cat8(*(const v8h*)&sK[cur][key][ 0 + half * 8],
                           *(const v8h*)&sK[cur][key][16 + half * 8]);
            v16h b1 = cat8(*(const v8h*)&sK[cur][key][32 + half * 8],
                           *(const v8h*)&sK[cur][key][48 + half * 8]);
            v8f c = {};
            c = __builtin_amdgcn_wmma_f32_16x16x32_f16(false, qa0, false, b0,
                                                       (short)0, c, false, false);
            c = __builtin_amdgcn_wmma_f32_16x16x32_f16(false, qa1, false, b1,
                                                       (short)0, c, false, false);
            sc[nb] = c;
        }

        // ---- row max: f32 local 4-way max, then packed-f16 permlane16 reduce ----
        // (f16 rounding of the max is safe: m is applied consistently to p and l,
        //  and can under-shoot the true max by at most ~0.015 in log2 domain)
        float mxf[8];
        #pragma unroll
        for (int i = 0; i < 8; ++i)
            mxf[i] = fmaxf(fmaxf(sc[0][i], sc[1][i]), fmaxf(sc[2][i], sc[3][i]));
        v2h hm[4];
        #pragma unroll
        for (int r = 0; r < 4; ++r) {
            hm[r][0] = (_Float16)mxf[2 * r];
            hm[r][1] = (_Float16)mxf[2 * r + 1];
        }
        #pragma unroll
        for (int r = 0; r < 4; ++r)
            hm[r] = __builtin_elementwise_max(hm[r], plx16h2(hm[r], 0x67452301u, 0xEFCDAB89u));
        #pragma unroll
        for (int r = 0; r < 4; ++r)
            hm[r] = __builtin_elementwise_max(hm[r], plx16h2(hm[r], 0x54761032u, 0xDCFE98BAu));
        #pragma unroll
        for (int r = 0; r < 4; ++r)
            hm[r] = __builtin_elementwise_max(hm[r], plx16h2(hm[r], 0x32107654u, 0xBA98FEDCu));
        #pragma unroll
        for (int r = 0; r < 4; ++r)
            hm[r] = __builtin_elementwise_max(hm[r], plx16h2(hm[r], 0xFEDCBA98u, 0x76543210u));

        // ---- online softmax (log2 domain) ----
        float alpha[8];
        #pragma unroll
        for (int i = 0; i < 8; ++i) {
            const float mnew = fmaxf(mrow[i], (float)hm[i >> 1][i & 1]);
            alpha[i] = __builtin_amdgcn_exp2f(mrow[i] - mnew);
            mrow[i]  = mnew;
            #pragma unroll
            for (int nb = 0; nb < 4; ++nb)
                sc[nb][i] = __builtin_amdgcn_exp2f(sc[nb][i] - mnew);
        }
        #pragma unroll
        for (int g = 0; g < 5; ++g)
            #pragma unroll
            for (int i = 0; i < 8; ++i) acc[g][i] *= alpha[i];

        // ---- P -> wave-private LDS with permuted columns j = 4*ln + nb ----
        #pragma unroll
        for (int i = 0; i < 8; ++i) {
            v4h p;
            #pragma unroll
            for (int nb = 0; nb < 4; ++nb) p[nb] = (_Float16)sc[nb][i];
            *(v4h*)&sP[wv][i + 8 * half][4 * ln] = p;
        }
        v16h pa0 = cat8(*(const v8h*)&sP[wv][ln][     kg],
                        *(const v8h*)&sP[wv][ln][16 + kg]);
        v16h pa1 = cat8(*(const v8h*)&sP[wv][ln][32 + kg],
                        *(const v8h*)&sP[wv][ln][48 + kg]);

        // ---- O += P V (permutation-consistent V^T); row sums via ones-B ----
        #pragma unroll
        for (int g = 0; g < 4; ++g) {
            v16h bv0 = cat8(*(const v8h*)&sVt[cur][g * 16 + ln][     half * 8],
                            *(const v8h*)&sVt[cur][g * 16 + ln][16 + half * 8]);
            v16h bv1 = cat8(*(const v8h*)&sVt[cur][g * 16 + ln][32 + half * 8],
                            *(const v8h*)&sVt[cur][g * 16 + ln][48 + half * 8]);
            acc[g] = __builtin_amdgcn_wmma_f32_16x16x32_f16(false, pa0, false, bv0,
                                                            (short)0, acc[g], false, false);
            acc[g] = __builtin_amdgcn_wmma_f32_16x16x32_f16(false, pa1, false, bv1,
                                                            (short)0, acc[g], false, false);
        }
        acc[4] = __builtin_amdgcn_wmma_f32_16x16x32_f16(false, pa0, false, bones,
                                                        (short)0, acc[4], false, false);
        acc[4] = __builtin_amdgcn_wmma_f32_16x16x32_f16(false, pa1, false, bones,
                                                        (short)0, acc[4], false, false);

        // ---- stage next tile into the other buffer, then signal ----
        if (more) {
            const int nxt = cur ^ 1;
            v8h a, b;
            const float* kr = (const float*)krf;
            #pragma unroll
            for (int j = 0; j < 8; ++j) { a[j] = (_Float16)kr[j]; b[j] = (_Float16)kr[8 + j]; }
            *(v8h*)&sK[nxt][srow][scol]     = a;
            *(v8h*)&sK[nxt][srow][scol + 8] = b;
            #pragma unroll
            for (int di = 0; di < 4; ++di) {
                v4h p;
                #pragma unroll
                for (int m = 0; m < 4; ++m) p[m] = (_Float16)((const float*)&vrf[m])[di];
                *(v4h*)&sVt[nxt][vd + di][4 * vc] = p;
            }
            asm volatile("s_wait_dscnt 0x0\n\ts_barrier_signal -1" ::: "memory");
        }
    }

    // ---- epilogue: divide by row sums (acc[4]), store f32 ----
    float* op = o + ((size_t)(h * S_ + qb + wv * 16)) * D_;
    #pragma unroll
    for (int i = 0; i < 8; ++i) {
        const float inv = 1.0f / acc[4][i];
        const int   row = i + 8 * half;
        #pragma unroll
        for (int g = 0; g < 4; ++g)
            op[(size_t)row * D_ + g * 16 + ln] = acc[g][i] * inv;
    }
}

extern "C" void kernel_launch(void* const* d_in, const int* in_sizes, int n_in,
                              void* d_out, int out_size, void* d_ws, size_t ws_size,
                              hipStream_t stream) {
    (void)in_sizes; (void)n_in; (void)out_size; (void)d_ws; (void)ws_size;
    const float* q = (const float*)d_in[0];
    const float* k = (const float*)d_in[1];
    const float* v = (const float*)d_in[2];
    // d_in[3] (mask) unused: reference discards the masked result.
    float* out = (float*)d_out;
    dim3 grid(S_ / ROWS_PER_BLOCK, H_);
    fa_fwd_gfx1250<<<grid, 256, 0, stream>>>(q, k, v, out);
}
